// QuantizedLinear_34746285425373
// MI455X (gfx1250) — matrix operand order repaired and checked by reading, not compiled, not run on hardware
//
#include <hip/hip_runtime.h>

// ---- problem constants (match reference setup_inputs) ----
#define M_TOTAL 8192              // B*S = 4*2048
#define K_DIM   4096
#define O_DIM   11008
#define GROUP   128               // quant group size == K block

// ---- tiling ----
#define BLOCK 512                 // 16 waves: 4 along M x 4 along N
#define MT   128                  // M rows per workgroup
#define NT   256                  // N cols per workgroup
#define KBLK 128                  // K per outer iteration (== GROUP)
#define LDSS 136                  // 128 + 8 halves pad -> 272B row stride (b128 bank-conflict free)

typedef _Float16 v16h __attribute__((ext_vector_type(16)));
typedef _Float16 v8h  __attribute__((ext_vector_type(8)));
typedef _Float16 v2h  __attribute__((ext_vector_type(2)));
typedef __fp16   v2hf __attribute__((ext_vector_type(2)));   // cvt_pkrtz return type
typedef float    v8f  __attribute__((ext_vector_type(8)));

// K-ordering: within every 8-value K group, LDS position p holds original
// k = [0,4,2,6,1,5,3,7][p]. Applied identically to the x tile and the W tile,
// so the WMMA dot product is unchanged; lets int4 dequant be pure mask/shift
// packed-f16 math with zero byte shuffles.

__global__ __launch_bounds__(BLOCK)
void qlinear_int4_wmma(const float* __restrict__ x,
                       const unsigned char* __restrict__ wq,   // [O, K/2] packed int4
                       const float* __restrict__ ws,           // [O, K/GROUP]
                       float* __restrict__ out)                // [M, O]
{
    extern __shared__ _Float16 smem[];
    _Float16* xLds = smem;                 // MT x LDSS halves
    _Float16* wLds = smem + MT * LDSS;     // NT x LDSS halves

    const int tid   = threadIdx.x;
    const int lane  = tid & 31;
    const int wave  = tid >> 5;          // 0..15
    const int l16   = lane & 15;
    const int lh    = lane >> 4;         // 0 or 1
    const int waveM = wave & 3;          // 4 waves along M, 32 rows each
    const int waveN = wave >> 2;         // 4 waves along N, 64 cols each

    const int mBase = blockIdx.x * MT;
    const int nBase = blockIdx.y * NT;

    v8f acc[2][4];
    #pragma unroll
    for (int a = 0; a < 2; ++a)
        #pragma unroll
        for (int j = 0; j < 4; ++j)
            acc[a][j] = (v8f){0.f,0.f,0.f,0.f,0.f,0.f,0.f,0.f};

    const float4* xv = reinterpret_cast<const float4*>(x);
    const uint4*  wv = reinterpret_cast<const uint4*>(wq);

    const v2h c1032 = {(_Float16)1032.0f, (_Float16)1032.0f};

    for (int kb = 0; kb < K_DIM / KBLK; ++kb) {
        // ---- stage x tile: 128 rows x 128 k, f32 -> f16 (permuted K) ----
        // 2048 groups of 8 floats, 4 groups per thread
        #pragma unroll
        for (int i = 0; i < 4; ++i) {
            int g   = tid + BLOCK * i;        // 0..2047
            int row = g >> 4;                 // 0..127
            int kg  = g & 15;                 // 8-float group within row
            size_t b4 = (size_t)(mBase + row) * (K_DIM / 4) + kb * (KBLK / 4) + kg * 2;
            float4 f0 = xv[b4];               // k0..k3
            float4 f1 = xv[b4 + 1];           // k4..k7
            v2hf p0 = __builtin_amdgcn_cvt_pkrtz(f0.x, f1.x);   // {k0,k4}
            v2hf p1 = __builtin_amdgcn_cvt_pkrtz(f0.z, f1.z);   // {k2,k6}
            v2hf p2 = __builtin_amdgcn_cvt_pkrtz(f0.y, f1.y);   // {k1,k5}
            v2hf p3 = __builtin_amdgcn_cvt_pkrtz(f0.w, f1.w);   // {k3,k7}
            uint4 o;
            o.x = __builtin_bit_cast(unsigned int, p0);
            o.y = __builtin_bit_cast(unsigned int, p1);
            o.z = __builtin_bit_cast(unsigned int, p2);
            o.w = __builtin_bit_cast(unsigned int, p3);
            *reinterpret_cast<uint4*>(&xLds[row * LDSS + kg * 8]) = o;
        }

        // ---- stage W tile: 256 rows x 128 k int4 -> f16 (permuted K) ----
        // 16KB packed = 1024 uint4, 2 per thread; one scale per row per K-block
        #pragma unroll
        for (int i = 0; i < 2; ++i) {
            int idx = tid + BLOCK * i;        // 0..1023
            int row = idx >> 2;               // 0..255
            int ch4 = idx & 3;                // 16B chunk (32 values)
            uint4 p = wv[(size_t)(nBase + row) * (K_DIM / 32) + kb * 4 + ch4];
            float s = ws[(size_t)(nBase + row) * (K_DIM / GROUP) + kb];
            _Float16 sh = (_Float16)s;
            v2h sv = {sh, sh};
            unsigned int wrd[4] = {p.x, p.y, p.z, p.w};
            #pragma unroll
            for (int wi = 0; wi < 4; ++wi) {
                unsigned int w0 = wrd[wi];
                // 0x6400|n is f16(1024+n), exact; (h-1032)*s : sub exact, mul rounds once
                unsigned int t0 = ( w0         & 0x000F000Fu) | 0x64006400u; // {k0,k4}
                unsigned int t1 = ((w0 >>  8)  & 0x000F000Fu) | 0x64006400u; // {k2,k6}
                unsigned int t2 = ((w0 >>  4)  & 0x000F000Fu) | 0x64006400u; // {k1,k5}
                unsigned int t3 = ((w0 >> 12)  & 0x000F000Fu) | 0x64006400u; // {k3,k7}
                v2h h0 = (__builtin_bit_cast(v2h, t0) - c1032) * sv;
                v2h h1 = (__builtin_bit_cast(v2h, t1) - c1032) * sv;
                v2h h2 = (__builtin_bit_cast(v2h, t2) - c1032) * sv;
                v2h h3 = (__builtin_bit_cast(v2h, t3) - c1032) * sv;
                uint4 o;
                o.x = __builtin_bit_cast(unsigned int, h0);
                o.y = __builtin_bit_cast(unsigned int, h1);
                o.z = __builtin_bit_cast(unsigned int, h2);
                o.w = __builtin_bit_cast(unsigned int, h3);
                *reinterpret_cast<uint4*>(&wLds[row * LDSS + ch4 * 32 + wi * 8]) = o;
            }
        }

        __syncthreads();

        // ---- WMMA: 4 k-steps of 32; 2 A-frags x 4 B-frags per wave ----
        #pragma unroll
        for (int kk = 0; kk < KBLK; kk += 32) {
            union { v16h v; v8h h[2]; } A0, A1;
            const _Float16* ar0 = &xLds[(waveM * 32 + l16) * LDSS + kk];
            const _Float16* ar1 = ar0 + 16 * LDSS;
            A0.h[0] = *reinterpret_cast<const v8h*>(ar0 + 8 * lh);
            A0.h[1] = *reinterpret_cast<const v8h*>(ar0 + 16 + 8 * lh);
            A1.h[0] = *reinterpret_cast<const v8h*>(ar1 + 8 * lh);
            A1.h[1] = *reinterpret_cast<const v8h*>(ar1 + 16 + 8 * lh);

            #pragma unroll
            for (int j = 0; j < 4; ++j) {
                union { v16h v; v8h h[2]; } Bf;
                const _Float16* br =
                    &wLds[(waveN * 64 + j * 16 + l16) * LDSS + kk + 8 * lh];
                Bf.h[0] = *reinterpret_cast<const v8h*>(br);
                Bf.h[1] = *reinterpret_cast<const v8h*>(br + 16);

                acc[0][j] = __builtin_amdgcn_wmma_f32_16x16x32_f16(
                    false, A0.v, false, Bf.v, (short)0, acc[0][j], false, false);
                acc[1][j] = __builtin_amdgcn_wmma_f32_16x16x32_f16(
                    false, A1.v, false, Bf.v, (short)0, acc[1][j], false, false);
            }
        }

        __syncthreads();
    }

    // ---- write out: D layout -> row = r + 8*(lane/16), col = lane%16 ----
    #pragma unroll
    for (int a = 0; a < 2; ++a) {
        #pragma unroll
        for (int j = 0; j < 4; ++j) {
            int col = nBase + waveN * 64 + j * 16 + l16;
            #pragma unroll
            for (int r = 0; r < 8; ++r) {
                int row = mBase + waveM * 32 + a * 16 + r + 8 * lh;
                out[(size_t)row * O_DIM + col] = acc[a][j][r];
            }
        }
    }
}

extern "C" void kernel_launch(void* const* d_in, const int* in_sizes, int n_in,
                              void* d_out, int out_size, void* d_ws, size_t ws_size,
                              hipStream_t stream) {
    (void)in_sizes; (void)n_in; (void)out_size; (void)d_ws; (void)ws_size;
    const float*         x  = (const float*)d_in[0];
    const unsigned char* wq = (const unsigned char*)d_in[1];
    const float*         ws = (const float*)d_in[2];
    float*               y  = (float*)d_out;

    const int ldsBytes = (MT + NT) * LDSS * (int)sizeof(_Float16);  // 104448
    (void)hipFuncSetAttribute(reinterpret_cast<const void*>(&qlinear_int4_wmma),
                              hipFuncAttributeMaxDynamicSharedMemorySize, ldsBytes);

    dim3 grid(M_TOTAL / MT, O_DIM / NT);   // 64 x 43
    dim3 block(BLOCK);
    qlinear_int4_wmma<<<grid, block, ldsBytes, stream>>>(x, wq, ws, y);
}
